// LabelWiseAttn_66718021976382
// MI455X (gfx1250) — matrix operand order repaired and checked by reading, not compiled, not run on hardware
//
#include <hip/hip_runtime.h>

// ---------------------------------------------------------------------------
// LabelWiseAttn fused kernel for MI455X (gfx1250, wave32, WMMA bf16)
//   att = U @ x^T ; alpha = softmax_L(att) ; m = alpha @ x
// ---------------------------------------------------------------------------

#define BB 16
#define LL 2048
#define DD 512
#define CC 8921
#define CPAD 8928          // 558 * 16
#define CTILES 558

typedef __attribute__((ext_vector_type(8)))  float  v8f;
typedef __attribute__((ext_vector_type(4)))  __bf16 v4bf;
typedef __attribute__((ext_vector_type(8)))  __bf16 v8bf;
typedef __attribute__((ext_vector_type(16))) __bf16 v16bf;

static __device__ __forceinline__ __bf16 f2bf(float f) {
  union { float f; unsigned u; } v; v.f = f;
  unsigned r = (v.u + 0x7FFFu + ((v.u >> 16) & 1u)) >> 16;   // RNE
  union { unsigned short s; __bf16 b; } o; o.s = (unsigned short)r;
  return o.b;
}

static __device__ __forceinline__ v16bf cat16(v8bf a, v8bf b) {
  return __builtin_shufflevector(a, b, 0,1,2,3,4,5,6,7,8,9,10,11,12,13,14,15);
}

// --- convert x: fp32 [B,L,D] -> bf16 row-major xb [B,L,D] and bf16 xT [B,D,L]
__global__ __launch_bounds__(256)
void k_convert_x(const float* __restrict__ x, __bf16* __restrict__ xb,
                 __bf16* __restrict__ xtb) {
  size_t gid = (size_t)blockIdx.x * blockDim.x + threadIdx.x;  // one per 4 elems
  size_t e = gid * 4;                                           // e < B*L*D
  float4 v = ((const float4*)x)[gid];
  float f[4] = {v.x, v.y, v.z, v.w};
  v4bf o; o[0]=f2bf(f[0]); o[1]=f2bf(f[1]); o[2]=f2bf(f[2]); o[3]=f2bf(f[3]);
  *(v4bf*)(xb + e) = o;
  size_t b = e / ((size_t)LL * DD);
  size_t rem = e % ((size_t)LL * DD);
  size_t l = rem / DD, d = rem % DD;
  #pragma unroll
  for (int i = 0; i < 4; ++i)
    xtb[(b * DD + d + i) * LL + l] = o[i];
}

// --- convert U: fp32 [C,D] -> bf16 [CPAD,D], zero-padded rows
__global__ __launch_bounds__(256)
void k_convert_U(const float* __restrict__ U, __bf16* __restrict__ ub) {
  size_t gid = (size_t)blockIdx.x * blockDim.x + threadIdx.x;  // one per 4 elems
  size_t e = gid * 4;                                           // e < CPAD*DD
  size_t c = e / DD;
  v4bf o;
  if (c < CC) {
    float4 v = ((const float4*)U)[gid];
    o[0]=f2bf(v.x); o[1]=f2bf(v.y); o[2]=f2bf(v.z); o[3]=f2bf(v.w);
  } else {
    o[0]=f2bf(0.f); o[1]=f2bf(0.f); o[2]=f2bf(0.f); o[3]=f2bf(0.f);
  }
  *(v4bf*)(ub + e) = o;
}

// --- fused main kernel: one block = (batch b, 16-label tile)
__global__ __launch_bounds__(256)
void lwa_main(const __bf16* __restrict__ xb,   // [B,L,D] bf16
              const __bf16* __restrict__ xtb,  // [B,D,L] bf16
              const __bf16* __restrict__ ub,   // [CPAD,D] bf16
              float* __restrict__ out) {       // [B,C,D] f32
  extern __shared__ char smem[];
  float*  att   = (float*)smem;                          // 16*2048 f32 = 128 KB
  __bf16* alpha = (__bf16*)(smem + 16*LL*4);             // 16*2048 bf16 = 64 KB
  float*  red   = (float*)(smem + 16*LL*4 + 16*LL*2);    // 16*16 f32
  float*  rowst = red + 256;                             // 16 f32

  const int ct   = blockIdx.x;         // label tile
  const int bb   = blockIdx.y;         // batch
  const int tid  = threadIdx.x;
  const int wave = tid >> 5;
  const int lane = tid & 31;
  const int lh   = lane >> 4;          // half-select (K group)
  const int lm   = lane & 15;          // M / N index within tile

  const int c_base = ct * 16;
  const __bf16* xrow  = xb  + (size_t)bb * LL * DD;
  const __bf16* xtrow = xtb + (size_t)bb * DD * LL;
  const __bf16* Urow  = ub  + (size_t)(c_base + lm) * DD;   // A row for this lane

  // ---------------- Phase 1: att[16, 2048] = U_tile @ x^T ----------------
  // wave w owns l in [w*256, w*256+256): 16 N-tiles of 16
  for (int t = 0; t < 16; ++t) {
    const int l_base = (wave * 16 + t) * 16;
    const __bf16* Xr = xrow + (size_t)(l_base + lm) * DD;    // B column row
    v8f acc = {};
    #pragma unroll
    for (int dc = 0; dc < 16; ++dc) {
      const int d0 = dc * 32;
      // A: 16x32 bf16, lane holds K = {lh*8+0..7, 16+lh*8+0..7}
      v8bf a0 = *(const v8bf*)(Urow + d0 + lh * 8);
      v8bf a1 = *(const v8bf*)(Urow + d0 + 16 + lh * 8);
      // B: 32x16 bf16, lane holds K = {lh*8+0..7, 16+lh*8+0..7} of column l_base+lm
      v8bf b0 = *(const v8bf*)(Xr + d0 + lh * 8);
      v8bf b1 = *(const v8bf*)(Xr + d0 + 16 + lh * 8);
      acc = __builtin_amdgcn_wmma_f32_16x16x32_bf16(
              false, cat16(a0, a1), false, cat16(b0, b1),
              (short)0, acc, false, false);
    }
    #pragma unroll
    for (int r = 0; r < 8; ++r)                 // D layout: M = r + 8*lh, N = lm
      att[(size_t)(r + 8 * lh) * LL + l_base + lm] = acc[r];
  }
  __syncthreads();

  // ---------------- Phase 2: softmax over L, write bf16 alpha -------------
  const int row  = tid >> 4;     // 0..15
  const int colg = tid & 15;     // 16 threads per row
  float mx = -3.402823466e38f;
  for (int j = colg; j < LL; j += 16) mx = fmaxf(mx, att[row * LL + j]);
  red[row * 16 + colg] = mx;
  __syncthreads();
  if (colg == 0) {
    float m2 = red[row * 16];
    #pragma unroll
    for (int i = 1; i < 16; ++i) m2 = fmaxf(m2, red[row * 16 + i]);
    rowst[row] = m2;
  }
  __syncthreads();
  const float rmax = rowst[row];
  float s = 0.f;
  for (int j = colg; j < LL; j += 16) {
    float e = __expf(att[row * LL + j] - rmax);
    att[row * LL + j] = e;
    s += e;
  }
  __syncthreads();
  red[row * 16 + colg] = s;
  __syncthreads();
  if (colg == 0) {
    float s2 = 0.f;
    #pragma unroll
    for (int i = 0; i < 16; ++i) s2 += red[row * 16 + i];
    rowst[row] = 1.0f / s2;
  }
  __syncthreads();
  const float rinv = rowst[row];
  for (int j = colg; j < LL; j += 16)
    alpha[row * LL + j] = f2bf(att[row * LL + j] * rinv);
  __syncthreads();

  // ---------------- Phase 3: m[16, 512] = alpha @ x -----------------------
  // wave w owns d in [w*64, w*64+64): 4 N-tiles of 16
  #pragma unroll
  for (int dt = 0; dt < 4; ++dt) {
    const int d_base = (wave * 4 + dt) * 16;
    const __bf16* XTr = xtrow + (size_t)(d_base + lm) * LL;  // B column (d) row in x^T
    v8f acc = {};
    #pragma unroll 4
    for (int lc = 0; lc < 64; ++lc) {
      const int l0 = lc * 32;
      // A: alpha row lm, K = {lh*8+0..7, 16+lh*8+0..7}  (ds_load_b128 from LDS)
      v8bf a0 = *(const v8bf*)(alpha + (size_t)lm * LL + l0 + lh * 8);
      v8bf a1 = *(const v8bf*)(alpha + (size_t)lm * LL + l0 + 16 + lh * 8);
      // B: x^T column d_base+lm, K = {lh*8+0..7, 16+lh*8+0..7}
      v8bf b0 = *(const v8bf*)(XTr + l0 + lh * 8);
      v8bf b1 = *(const v8bf*)(XTr + l0 + 16 + lh * 8);
      acc = __builtin_amdgcn_wmma_f32_16x16x32_bf16(
              false, cat16(a0, a1), false, cat16(b0, b1),
              (short)0, acc, false, false);
    }
    #pragma unroll
    for (int r = 0; r < 8; ++r) {
      const int c = c_base + r + 8 * lh;
      if (c < CC)
        out[((size_t)bb * CC + c) * DD + d_base + lm] = acc[r];
    }
  }
}

extern "C" void kernel_launch(void* const* d_in, const int* in_sizes, int n_in,
                              void* d_out, int out_size, void* d_ws, size_t ws_size,
                              hipStream_t stream) {
  const float* x = (const float*)d_in[0];     // [B,L,D] f32
  const float* U = (const float*)d_in[1];     // [C,D]  f32
  float* out = (float*)d_out;                 // [B,C,D] f32

  char* ws = (char*)d_ws;
  const size_t xb_bytes  = (size_t)BB * LL * DD * 2;   // 32 MiB
  const size_t xtb_bytes = (size_t)BB * DD * LL * 2;   // 32 MiB
  __bf16* xb  = (__bf16*)ws;
  __bf16* xtb = (__bf16*)(ws + xb_bytes);
  __bf16* ub  = (__bf16*)(ws + xb_bytes + xtb_bytes);  // CPAD*DD*2 ~ 8.7 MiB

  // precision/layout prep: fp32 -> bf16, plus transposed copy of x
  const int nx4 = BB * LL * DD / 4;           // 4,194,304 threads
  k_convert_x<<<nx4 / 256, 256, 0, stream>>>(x, xb, xtb);
  const int nu4 = CPAD * DD / 4;              // 1,142,784 threads
  k_convert_U<<<nu4 / 256, 256, 0, stream>>>(U, ub);

  const size_t smem = (size_t)16 * LL * 4     // att  f32
                    + (size_t)16 * LL * 2     // alpha bf16
                    + 256 * 4 + 16 * 4;       // reduction scratch
  (void)hipFuncSetAttribute((const void*)lwa_main,
                            hipFuncAttributeMaxDynamicSharedMemorySize,
                            (int)smem);
  dim3 grid(CTILES, BB);
  lwa_main<<<grid, 256, smem, stream>>>(xb, xtb, ub, out);
}
